// DynamicMHC_69200513073696
// MI455X (gfx1250) — matrix operand order repaired and checked by reading, not compiled, not run on hardware
//
#include <hip/hip_runtime.h>
#include <hip/hip_bf16.h>
#include <cstdint>

typedef __attribute__((ext_vector_type(16))) __bf16        v16bf;
typedef __attribute__((ext_vector_type(8)))  float         v8f;
typedef __attribute__((ext_vector_type(8)))  unsigned int  v8u;

#define N_STREAMS   4
#define CDIM        1024
#define DDIM        4096
#define TOK_PER_WG  64
#define THREADS     256
#define KCH         128
#define NKCH        (DDIM / KCH)     // 32 chunks
#define TAU_INV     10.0f
#define SINK_ITERS  50
#define RMS_EPS     1.1920929e-07f

// Manual RNE f32->bf16 pack: lowers to plain 32-bit VALU ops; unlike
// __builtin_amdgcn_cvt_pk_bf16_f32 (which legalizes to 16-bit subreg
// converts + half-register moves on gfx1250).
__device__ __forceinline__ unsigned int pack2bf(float a, float b) {
    unsigned int ua = __float_as_uint(a);
    unsigned int ub = __float_as_uint(b);
    ua = (ua + 0x7FFFu + ((ua >> 16) & 1u)) >> 16;     // RNE
    ub = (ub + 0x7FFFu + ((ub >> 16) & 1u)) >> 16;
    return ua | (ub << 16);
}
__device__ __forceinline__ float grp_max4(float v) {   // max over lane groups of 4
    v = fmaxf(v, __shfl_xor(v, 1));
    v = fmaxf(v, __shfl_xor(v, 2));
    return v;
}
__device__ __forceinline__ float grp_sum4(float v) {   // sum over lane groups of 4
    v += __shfl_xor(v, 1);
    v += __shfl_xor(v, 2);
    return v;
}

__global__ __launch_bounds__(THREADS)
void mhc_fused_kernel(const float* __restrict__ x,
                      const float* __restrict__ Wres,
                      const float* __restrict__ Wpre,
                      const float* __restrict__ Wpost,
                      const float* __restrict__ HresB,
                      const float* __restrict__ HpreB,
                      const float* __restrict__ HpostB,
                      const float* __restrict__ gate,
                      float* __restrict__ out)
{
    __shared__ __align__(16) unsigned short sA[TOK_PER_WG * KCH]; // x_norm bf16 tile (16 KB)
    __shared__ __align__(16) unsigned short sW[32 * KCH];         // weights bf16, rows 24..31 = 0 (8 KB)
    __shared__ float sLog[TOK_PER_WG][32];                        // projection logits (8 KB)
    __shared__ float sRstd[TOK_PER_WG];
    __shared__ float sH[TOK_PER_WG][24];                          // Hres(16) | Hpre(4) | Hpost(4)

    const int tid  = threadIdx.x;
    const int lane = tid & 31;
    const int wave = tid >> 5;                 // 8 waves (wave32)
    const int blockBase = blockIdx.x * TOK_PER_WG;

    // ---------- Phase 0: RMSNorm scale (b128 loads, wave-reduce) ----------
    for (int t = 0; t < 8; ++t) {
        int tk = wave * 8 + t;
        const float4* row = reinterpret_cast<const float4*>(
            x + (size_t)(blockBase + tk) * DDIM);
        float s = 0.f;
        #pragma unroll 4
        for (int j = lane; j < DDIM / 4; j += 32) {
            float4 v = row[j];
            s = fmaf(v.x, v.x, s);
            s = fmaf(v.y, v.y, s);
            s = fmaf(v.z, v.z, s);
            s = fmaf(v.w, v.w, s);
        }
        #pragma unroll
        for (int m = 16; m; m >>= 1) s += __shfl_xor(s, m);
        if (lane == 0) sRstd[tk] = rsqrtf(s * (1.0f / DDIM) + RMS_EPS);
    }
    __syncthreads();

    // ---------- Phase 1: WMMA projection  logits = x_norm @ W^T ----------
    const int nt  = wave & 1;     // output-column tile: 0..15 / 16..31
    const int mt  = wave >> 1;    // token-row tile: mt*16 .. mt*16+15
    const int rlo = lane & 15;
    const int kh  = lane >> 4;
    v8f acc = {};

    for (int kc = 0; kc < NKCH; ++kc) {
        // stage x_norm chunk: 64 tokens x 128 k as bf16
        {
            int tk = tid >> 2;
            int ks = (tid & 3) * 32;
            float rs = sRstd[tk];
            const float4* src = reinterpret_cast<const float4*>(
                x + (size_t)(blockBase + tk) * DDIM + kc * KCH + ks);
            unsigned int* dst = reinterpret_cast<unsigned int*>(&sA[tk * KCH + ks]);
            #pragma unroll
            for (int q = 0; q < 8; ++q) {
                float4 v = src[q];
                dst[2 * q + 0] = pack2bf(v.x * rs, v.y * rs);
                dst[2 * q + 1] = pack2bf(v.z * rs, v.w * rs);
            }
            if (kc + 1 < NKCH)   // global_prefetch_b8 for the next K chunk
                __builtin_prefetch(
                    x + (size_t)(blockBase + tk) * DDIM + (kc + 1) * KCH + ks, 0, 1);
        }
        // stage weight chunk: rows 0..15 W_res, 16..19 W_pre, 20..23 W_post, 24..31 zero
        {
            int row = tid >> 3;
            int ks  = (tid & 7) * 16;
            unsigned int* dst = reinterpret_cast<unsigned int*>(&sW[row * KCH + ks]);
            const float* wrow = nullptr;
            if (row < 16)      wrow = Wres  + (size_t)row * DDIM;
            else if (row < 20) wrow = Wpre  + (size_t)(row - 16) * DDIM;
            else if (row < 24) wrow = Wpost + (size_t)(row - 20) * DDIM;
            if (wrow) {
                const float4* src = reinterpret_cast<const float4*>(wrow + kc * KCH + ks);
                #pragma unroll
                for (int q = 0; q < 4; ++q) {
                    float4 v = src[q];
                    dst[2 * q + 0] = pack2bf(v.x, v.y);
                    dst[2 * q + 1] = pack2bf(v.z, v.w);
                }
            } else {
                #pragma unroll
                for (int q = 0; q < 8; ++q) dst[q] = 0u;
            }
        }
        __syncthreads();

        // 4 x (K=32) WMMA steps over this 128-wide chunk
        const uint4* a16 = reinterpret_cast<const uint4*>(sA);
        const uint4* w16 = reinterpret_cast<const uint4*>(sW);
        #pragma unroll
        for (int kk = 0; kk < 4; ++kk) {
            int ks = kk * 32;
            // A fragment (ISA 7.12.2, 16-bit A 16x32): lanes 0-15 K0-7/K16-23, lanes 16-31 K8-15/K24-31
            int arow = mt * 16 + rlo;
            uint4 alo = a16[(arow * KCH + ks + 8 * kh) >> 3];
            uint4 ahi = a16[(arow * KCH + ks + 16 + 8 * kh) >> 3];
            // B fragment: lane = output column, same K split as A (lanes 0-15 K0-7/K16-23, lanes 16-31 K8-15/K24-31)
            int brow = nt * 16 + rlo;
            uint4 blo = w16[(brow * KCH + ks + 8 * kh) >> 3];
            uint4 bhi = w16[(brow * KCH + ks + 16 + 8 * kh) >> 3];
            v8u ar = { alo.x, alo.y, alo.z, alo.w, ahi.x, ahi.y, ahi.z, ahi.w };
            v8u br = { blo.x, blo.y, blo.z, blo.w, bhi.x, bhi.y, bhi.z, bhi.w };
            v16bf av = __builtin_bit_cast(v16bf, ar);
            v16bf bv = __builtin_bit_cast(v16bf, br);
            acc = __builtin_amdgcn_wmma_f32_16x16x32_bf16(false, av, false, bv,
                                                          (short)0, acc, false, false);
        }
        __syncthreads();
    }

    // ---------- Phase 2: C tile -> logits LDS (C layout: M=r+8*kh, N=lane&15) ----------
    #pragma unroll
    for (int r = 0; r < 8; ++r) {
        int m  = mt * 16 + r + 8 * kh;
        int nn = nt * 16 + rlo;
        sLog[m][nn] = acc[r];
    }
    __syncthreads();

    // ---------- Phase 3: Sinkhorn (4 lanes per token) + softmaxes ----------
    {
        int tk = tid >> 2;        // token 0..63
        int i  = tid & 3;         // row of the 4x4 problem
        float g = 1.0f / (1.0f + __expf(-gate[0]));
        float z[4];
        #pragma unroll
        for (int j = 0; j < 4; ++j)
            z[j] = (HresB[i * 4 + j] + sLog[tk][i * 4 + j]) * TAU_INV;
        float preL  = HpreB[i]  + sLog[tk][16 + i];
        float postL = HpostB[i] + sLog[tk][20 + i];

        float u = 0.f, v0 = 0.f, v1 = 0.f, v2 = 0.f, v3 = 0.f;
        for (int it = 0; it < SINK_ITERS; ++it) {
            // u_i = -logsumexp_j(Z[i][j] + v[j])   (lane-local)
            float t0 = z[0] + v0, t1 = z[1] + v1, t2 = z[2] + v2, t3 = z[3] + v3;
            float m = fmaxf(fmaxf(t0, t1), fmaxf(t2, t3));
            float s = __expf(t0 - m) + __expf(t1 - m) + __expf(t2 - m) + __expf(t3 - m);
            u = -(m + __logf(s));
            // v_j = -logsumexp_i(Z[i][j] + u_i)    (cross-lane within group of 4)
            #pragma unroll
            for (int j = 0; j < 4; ++j) {
                float t  = z[j] + u;
                float mm = grp_max4(t);
                float e  = __expf(t - mm);
                float vj = -(mm + __logf(grp_sum4(e)));
                if (j == 0) v0 = vj; else if (j == 1) v1 = vj;
                else if (j == 2) v2 = vj; else v3 = vj;
            }
        }
        float vv[4] = { v0, v1, v2, v3 };
        #pragma unroll
        for (int j = 0; j < 4; ++j) {
            float H = __expf(z[j] + u + vv[j]);
            sH[tk][i * 4 + j] = ((i == j) ? (1.0f - g) : 0.0f) + g * H;
        }
        { float m = grp_max4(preL);  float e = __expf(preL  - m); sH[tk][16 + i] = e / grp_sum4(e); }
        { float m = grp_max4(postL); float e = __expf(postL - m); sH[tk][20 + i] = e / grp_sum4(e); }
    }
    __syncthreads();

    // ---------- Phase 4: stream mixing, float4 wide (x re-read is L2-hot) ----------
    for (int idx = tid; idx < TOK_PER_WG * (CDIM / 4); idx += THREADS) {
        int tk = idx >> 8;                 // CDIM/4 = 256 quads per token
        int c4 = (idx & 255) * 4;
        const float* xt = x + (size_t)(blockBase + tk) * DDIM + c4;
        float xs[4][4];
        #pragma unroll
        for (int n = 0; n < 4; ++n) {
            float4 v = *reinterpret_cast<const float4*>(xt + n * CDIM);
            xs[n][0] = v.x; xs[n][1] = v.y; xs[n][2] = v.z; xs[n][3] = v.w;
        }
        const float* h = sH[tk];
        float* ot = out + (size_t)(blockBase + tk) * DDIM + c4;
        float ys[4];
        #pragma unroll
        for (int e = 0; e < 4; ++e)
            ys[e] = xs[0][e] * h[16] + xs[1][e] * h[17] +
                    xs[2][e] * h[18] + xs[3][e] * h[19];
        #pragma unroll
        for (int i = 0; i < 4; ++i) {
            float4 ov;
            float* oe = &ov.x;
            #pragma unroll
            for (int e = 0; e < 4; ++e)
                oe[e] = h[i * 4 + 0] * xs[0][e] + h[i * 4 + 1] * xs[1][e] +
                        h[i * 4 + 2] * xs[2][e] + h[i * 4 + 3] * xs[3][e] +
                        ys[e] * h[20 + i];
            *reinterpret_cast<float4*>(ot + i * CDIM) = ov;
        }
    }
}

extern "C" void kernel_launch(void* const* d_in, const int* in_sizes, int n_in,
                              void* d_out, int out_size, void* d_ws, size_t ws_size,
                              hipStream_t stream) {
    (void)n_in; (void)out_size; (void)d_ws; (void)ws_size;
    const float* x     = (const float*)d_in[0];
    const float* Wres  = (const float*)d_in[1];
    const float* Wpre  = (const float*)d_in[2];
    const float* Wpost = (const float*)d_in[3];
    const float* HresB = (const float*)d_in[4];
    const float* HpreB = (const float*)d_in[5];
    const float* HpostB= (const float*)d_in[6];
    const float* gate  = (const float*)d_in[7];
    float* out = (float*)d_out;

    int tokens = in_sizes[0] / DDIM;            // B*T = 16384
    int grid   = tokens / TOK_PER_WG;           // 256 workgroups
    mhc_fused_kernel<<<grid, THREADS, 0, stream>>>(
        x, Wres, Wpre, Wpost, HresB, HpreB, HpostB, gate, out);
}
